// TaskAlignedAssigner_23467701305827
// MI455X (gfx1250) — matrix operand order repaired and checked by reading, not compiled, not run on hardware
//
#include <hip/hip_runtime.h>
#include <cstdint>
#include <cstddef>

// TaskAlignedAssigner for MI455X (gfx1250, wave32).
// Inputs (setup_inputs order): pd_scores f32[B,A,80], pd_bboxes f32[B,A,4],
// anc_points f32[A,2], gt_labels i32[B,64,1], gt_bboxes f32[B,64,4],
// mask_gt bool8[B,64,1].
// Outputs concatenated flat as f32: t_labels[B,A], t_bboxes[B,A,4],
// t_scores[B,A,80], fg[B,A].
//
// Workspace (assumed >= ~72 MB): amf f32[B,64,A] (GT-major: coalesced for both
// column top-k and per-anchor loops), thr f32[B,64], am_max f32[B,A],
// gt_idx i32[B,A], fg i32[B,A], mnmx u32[B,2].

#define NUM_CLASSES 80
#define NGT 64
#define TOPK 10

typedef __attribute__((ext_vector_type(16))) _Float16 v16h;
typedef __attribute__((ext_vector_type(8)))  float    v8f;

__device__ __forceinline__ void topk_insert(float v, float (&t)[TOPK]) {
  if (v <= t[TOPK - 1]) return;
  t[TOPK - 1] = v;
#pragma unroll
  for (int j = TOPK - 1; j > 0; --j) {
    if (t[j] > t[j - 1]) { float tmp = t[j - 1]; t[j - 1] = t[j]; t[j] = tmp; }
  }
}

// ---------------------------------------------------------------- K0: init
__global__ void k0_init(unsigned int* __restrict__ mnmx, int B) {
  int i = blockIdx.x * blockDim.x + threadIdx.x;
  if (i < B) {
    mnmx[2 * i]     = __float_as_uint(1e30f); // running min of am_max (>=0)
    mnmx[2 * i + 1] = 0u;                     // running max (0.0f)
  }
}

// ---------------------------------------------------------------- K1: amf
// scores[a,n] = pd_scores[a, labels[n]] computed as WMMA GEMM with one-hot B:
// per wave, a 16(anchor)x64(gt) f32 tile from 3x v_wmma_f32_16x16x32_f16
// k-steps (C=80 zero-padded to 96) x 4 n-tiles, then LDS transpose.
//
// A-operand fetch: per the 16-bit A 16x32 VGPR layout, a lane's slots across
// all 3 k-steps are the K runs {16j + 8*khalf .. +8}, j=0..4 -- all < 80.
// So: 10 unconditional float4 loads, one f32->f16 convert pass, and the three
// v16h operands are a compile-time rearrangement (K>=80 half is a static 0).
__global__ __launch_bounds__(128) void k1_amf(
    const float* __restrict__ pd_scores, const float* __restrict__ pd_bboxes,
    const float* __restrict__ anc, const int* __restrict__ gt_labels,
    const float* __restrict__ gt_bboxes, const unsigned char* __restrict__ mask_gt,
    float* __restrict__ ws_amf, float* __restrict__ ws_ammax,
    int* __restrict__ ws_gtidx, int A)
{
  const int b   = blockIdx.y;
  const int a0  = blockIdx.x * 128;
  const int tid = threadIdx.x;
  const int lane = tid & 31;
  const int wv   = tid >> 5;

  __shared__ float s_gt[NGT][4];
  __shared__ int   s_lab[NGT];
  __shared__ float s_val[NGT];
  __shared__ float s_sc[128][65];   // +1 pad -> conflict-free per-anchor row reads

  if (tid < NGT) {
    const float* g = gt_bboxes + ((size_t)b * NGT + tid) * 4;
    s_gt[tid][0] = g[0]; s_gt[tid][1] = g[1];
    s_gt[tid][2] = g[2]; s_gt[tid][3] = g[3];
    s_lab[tid] = gt_labels[(size_t)b * NGT + tid];
    s_val[tid] = mask_gt[(size_t)b * NGT + tid] ? 1.0f : 0.0f;
  }
  __syncthreads();

  const int mrow  = lane & 15;  // M row (A) / N col (B,C/D) within 16x16 tile
  const int khalf = lane >> 4;  // selects K sub-range per ISA VGPR layout

#pragma unroll
  for (int t = 0; t < 2; ++t) {               // each wave: 2 anchor tiles
    const int aBase = a0 + (wv * 2 + t) * 16;
    const int arow  = aBase + mrow;
    // Clamp row for loads (A % 16 == 0 -> OOB tiles never stored); keeps the
    // WMMA region free of EXEC-divergent branches.
    const int arowc = (arow < A) ? arow : (A - 1);
    const float* rowp =
        pd_scores + ((size_t)b * A + arowc) * NUM_CLASSES + khalf * 8;

    _Float16 h[48];                           // 5 runs of 8 + zero run (K>=80)
#pragma unroll
    for (int j = 0; j < 5; ++j) {             // K = 16j + 8*khalf .. +8, all <80
      const float4 lo = *(const float4*)(rowp + 16 * j);
      const float4 hi = *(const float4*)(rowp + 16 * j + 4);
      h[8 * j + 0] = (_Float16)lo.x; h[8 * j + 1] = (_Float16)lo.y;
      h[8 * j + 2] = (_Float16)lo.z; h[8 * j + 3] = (_Float16)lo.w;
      h[8 * j + 4] = (_Float16)hi.x; h[8 * j + 5] = (_Float16)hi.y;
      h[8 * j + 6] = (_Float16)hi.z; h[8 * j + 7] = (_Float16)hi.w;
    }
#pragma unroll
    for (int i = 40; i < 48; ++i) h[i] = (_Float16)0.0f;

    v8f acc[4];
#pragma unroll
    for (int nt = 0; nt < 4; ++nt)
#pragma unroll
      for (int r = 0; r < 8; ++r) acc[nt][r] = 0.0f;

#pragma unroll
    for (int kk = 0; kk < 3; ++kk) {          // K = 3*32 = 96 >= 80
      v16h am;                                // compile-time slot mapping
#pragma unroll
      for (int i = 0; i < 16; ++i)
        am[i] = h[8 * (2 * kk + (i >= 8 ? 1 : 0)) + (i & 7)];
#pragma unroll
      for (int nt = 0; nt < 4; ++nt) {
        // B operand: one-hot(label[n]==k); lanes0-15 hold K 0-7,16-23; 16-31: 8-15,24-31
        const int lbl = s_lab[nt * 16 + mrow];
        v16h bm;
#pragma unroll
        for (int i = 0; i < 16; ++i) {
          int k = 32 * kk + 8 * khalf + (i < 8 ? i : 16 + (i - 8));
          bm[i] = (lbl == k) ? (_Float16)1.0f : (_Float16)0.0f;
        }
        acc[nt] = __builtin_amdgcn_wmma_f32_16x16x32_f16(
            false, am, false, bm, (short)0, acc[nt], false, false);
      }
    }
    // D tile -> LDS transpose: lane holds rows M=r (lanes0-15) / M=r+8 (16-31)
#pragma unroll
    for (int nt = 0; nt < 4; ++nt)
#pragma unroll
      for (int r = 0; r < 8; ++r)
        s_sc[(aBase - a0) + r + khalf * 8][nt * 16 + mrow] = acc[nt][r];
  }
  __syncthreads();

  // Per-anchor: IoU, in_gts, align metric, row max/argmax; GT-major amf store.
  const int a = a0 + tid;
  if (a < A) {
    const float4 pb = ((const float4*)pd_bboxes)[(size_t)b * A + a];
    const float ax = anc[2 * a], ay = anc[2 * a + 1];
    float best = -2.0f; int bestg = 0;
    for (int g = 0; g < NGT; ++g) {
      const float gx1 = s_gt[g][0], gy1 = s_gt[g][1];
      const float gx2 = s_gt[g][2], gy2 = s_gt[g][3];
      float iw = fmaxf(fminf(gx2, pb.z) - fmaxf(gx1, pb.x), 0.0f);
      float ih = fmaxf(fminf(gy2, pb.w) - fmaxf(gy1, pb.y), 0.0f);
      float inter = iw * ih;
      float w1 = gx2 - gx1, h1 = gy2 - gy1 + 1e-16f;
      float w2 = pb.z - pb.x, h2 = pb.w - pb.y + 1e-16f;
      float iou = inter / (w1 * h1 + w2 * h2 - inter + 1e-16f);
      float sc  = fmaxf(s_sc[tid][g], 0.0f);
      float i2  = iou * iou;
      float align = sqrtf(sc) * (i2 * i2 * i2);
      bool ing = (ax - gx1 > 0.0f) && (ay - gy1 > 0.0f) &&
                 (gx2 - ax > 0.0f) && (gy2 - ay > 0.0f);
      float amf = (ing ? align : 0.0f) * s_val[g];
      ws_amf[((size_t)b * NGT + g) * A + a] = amf;   // coalesced over a
      float arg = (s_val[g] != 0.0f) ? amf : -1.0f;
      if (arg > best) { best = arg; bestg = g; }     // first-max == jnp.argmax
    }
    ws_ammax[(size_t)b * A + a] = fmaxf(best, 0.0f);
    ws_gtidx[(size_t)b * A + a] = bestg;
  }
}

// ------------------------------------------------- K2: per-GT top-10 threshold
__global__ __launch_bounds__(256) void k2_thr(const float* __restrict__ ws_amf,
                                              float* __restrict__ ws_thr, int A)
{
  const int g = blockIdx.x, b = blockIdx.y;
  const float* col = ws_amf + ((size_t)b * NGT + g) * A;
  const int tid = threadIdx.x;

  float t[TOPK];
#pragma unroll
  for (int j = 0; j < TOPK; ++j) t[j] = -1.0f;
  for (int i = tid; i < A; i += 256) topk_insert(col[i], t);  // coalesced

  __shared__ float s1[256 * TOPK];
  __shared__ float s2[32 * TOPK];
#pragma unroll
  for (int j = 0; j < TOPK; ++j) s1[tid * TOPK + j] = t[j];
  __syncthreads();
  if (tid < 32) {
    float u[TOPK];
#pragma unroll
    for (int j = 0; j < TOPK; ++j) u[j] = -1.0f;
    for (int i = 0; i < 8 * TOPK; ++i) topk_insert(s1[tid * 8 * TOPK + i], u);
#pragma unroll
    for (int j = 0; j < TOPK; ++j) s2[tid * TOPK + j] = u[j];
  }
  __syncthreads();
  if (tid == 0) {
    float u[TOPK];
#pragma unroll
    for (int j = 0; j < TOPK; ++j) u[j] = -1.0f;
    for (int i = 0; i < 32 * TOPK; ++i) topk_insert(s2[i], u);
    ws_thr[(size_t)b * NGT + g] = u[TOPK - 1];      // 10th largest of column
  }
}

// ----------------------------------------- K3: fg mask + per-batch min/max
__global__ __launch_bounds__(256) void k3_fg(const float* __restrict__ ws_amf,
    const float* __restrict__ ws_thr, const float* __restrict__ ws_ammax,
    int* __restrict__ ws_fg, unsigned int* __restrict__ ws_mnmx, int A)
{
  const int b = blockIdx.y;
  const int tid = threadIdx.x;
  const int a = blockIdx.x * 256 + tid;
  __shared__ float s_thr[NGT];
  if (tid < NGT) s_thr[tid] = ws_thr[(size_t)b * NGT + tid];
  __syncthreads();

  float mn = 1e30f, mx = 0.0f;   // identities (am_max >= 0)
  int fg = 0;
  if (a < A) {
    for (int g = 0; g < NGT; ++g) {
      float v = ws_amf[((size_t)b * NGT + g) * A + a];   // coalesced over a
      // amf>0 implies in_gts & valid; >=thr approximates membership in top-10
      // (exact except at equal-value ties; when thr==0, <10 positives exist and
      //  amf>0 selects exactly those).
      fg |= (int)(v > 0.0f && v >= s_thr[g]);
    }
    if (fg) { float am = ws_ammax[(size_t)b * A + a]; mn = am; mx = am; }
    ws_fg[(size_t)b * A + a] = fg;
  }
  __shared__ float smin[256], smax[256];
  smin[tid] = mn; smax[tid] = mx;
  __syncthreads();
  for (int s = 128; s > 0; s >>= 1) {
    if (tid < s) {
      smin[tid] = fminf(smin[tid], smin[tid + s]);
      smax[tid] = fmaxf(smax[tid], smax[tid + s]);
    }
    __syncthreads();
  }
  if (tid == 0) {
    // nonneg f32: bit pattern order == value order -> u32 atomics, deterministic
    atomicMin(&ws_mnmx[2 * b],     __float_as_uint(smin[0]));
    atomicMax(&ws_mnmx[2 * b + 1], __float_as_uint(smax[0]));
  }
}

// -------------------------------------------------------- K4: write outputs
__global__ __launch_bounds__(256) void k4_out(
    const int* __restrict__ gt_labels, const float* __restrict__ gt_bboxes,
    const float* __restrict__ ws_ammax, const int* __restrict__ ws_gtidx,
    const int* __restrict__ ws_fg, const unsigned int* __restrict__ ws_mnmx,
    float* __restrict__ o_lab, float* __restrict__ o_bb,
    float* __restrict__ o_sc, float* __restrict__ o_fg, int A)
{
  const int b = blockIdx.y;
  const int base = blockIdx.x * 256;
  const int tid = threadIdx.x;

  // coalesced zero-fill of this block's t_scores rows
  float* scRow = o_sc + ((size_t)b * A + base) * NUM_CLASSES;
  const int limit = min(256, A - base) * NUM_CLASSES;
  for (int i = tid; i < limit; i += 256) scRow[i] = 0.0f;
  __syncthreads();

  const int a = base + tid;
  if (a >= A) return;
  const size_t ba = (size_t)b * A + a;
  const int fg  = ws_fg[ba];
  const int gi  = ws_gtidx[ba];
  const int lbl = fg ? gt_labels[(size_t)b * NGT + gi] : NUM_CLASSES;
  o_lab[ba] = (float)lbl;
  o_fg[ba]  = fg ? 1.0f : 0.0f;
  float4 bb = make_float4(0.0f, 0.0f, 0.0f, 0.0f);
  if (fg) bb = ((const float4*)gt_bboxes)[(size_t)b * NGT + gi];
  ((float4*)o_bb)[ba] = bb;
  if (fg) {
    const float mn = __uint_as_float(ws_mnmx[2 * b]);
    const float mx = __uint_as_float(ws_mnmx[2 * b + 1]);
    const float norm = (ws_ammax[ba] - mn) / (mx - mn + 1e-9f);
    scRow[tid * NUM_CLASSES + lbl] = norm;   // one-hot * norm
  }
}

// ------------------------------------------------------------------- launch
extern "C" void kernel_launch(void* const* d_in, const int* in_sizes, int n_in,
                              void* d_out, int out_size, void* d_ws, size_t ws_size,
                              hipStream_t stream)
{
  (void)n_in; (void)out_size; (void)ws_size;
  const float* pd_scores = (const float*)d_in[0];
  const float* pd_bboxes = (const float*)d_in[1];
  const float* anc       = (const float*)d_in[2];
  const int*   gt_labels = (const int*)d_in[3];
  const float* gt_bboxes = (const float*)d_in[4];
  const unsigned char* mask_gt = (const unsigned char*)d_in[5]; // jax bool = 1 byte

  const int A = in_sizes[2] / 2;                    // 8400
  const int B = in_sizes[0] / (A * NUM_CLASSES);    // 32  (N fixed at 64 = NGT)

  float* ws_amf   = (float*)d_ws;                          // B*NGT*A
  float* ws_thr   = ws_amf + (size_t)B * NGT * A;          // B*NGT
  float* ws_ammax = ws_thr + (size_t)B * NGT;              // B*A
  int*   ws_gtidx = (int*)(ws_ammax + (size_t)B * A);      // B*A
  int*   ws_fg    = ws_gtidx + (size_t)B * A;              // B*A
  unsigned int* ws_mnmx = (unsigned int*)(ws_fg + (size_t)B * A); // 2*B

  float* out   = (float*)d_out;
  float* o_lab = out;
  float* o_bb  = o_lab + (size_t)B * A;
  float* o_sc  = o_bb + (size_t)B * A * 4;
  float* o_fg  = o_sc + (size_t)B * A * NUM_CLASSES;

  k0_init<<<dim3((B + 63) / 64), 64, 0, stream>>>(ws_mnmx, B);
  k1_amf<<<dim3((A + 127) / 128, B), 128, 0, stream>>>(
      pd_scores, pd_bboxes, anc, gt_labels, gt_bboxes, mask_gt,
      ws_amf, ws_ammax, ws_gtidx, A);
  k2_thr<<<dim3(NGT, B), 256, 0, stream>>>(ws_amf, ws_thr, A);
  k3_fg<<<dim3((A + 255) / 256, B), 256, 0, stream>>>(
      ws_amf, ws_thr, ws_ammax, ws_fg, ws_mnmx, A);
  k4_out<<<dim3((A + 255) / 256, B), 256, 0, stream>>>(
      gt_labels, gt_bboxes, ws_ammax, ws_gtidx, ws_fg, ws_mnmx,
      o_lab, o_bb, o_sc, o_fg, A);
}
